// HMM_42434276884836
// MI455X (gfx1250) — matrix operand order repaired and checked by reading, not compiled, not run on hardware
//
#include <hip/hip_runtime.h>
#include <hip/hip_bf16.h>
#include <math.h>

#define NSTATES 512
#define VOCABN  10000
#define TMAXN   1024

typedef __attribute__((ext_vector_type(16))) __bf16        v16bf;
typedef __attribute__((ext_vector_type(8)))  float         v8f;
typedef __attribute__((ext_vector_type(4)))  unsigned int  u32x4;

union ABu { v16bf v; u32x4 q[2]; };

__device__ __forceinline__ unsigned short f2bf(float f) {
    unsigned int u = __float_as_uint(f);
    unsigned int r = u + 0x7fffu + ((u >> 16) & 1u);   // round-to-nearest-even
    return (unsigned short)(r >> 16);
}

__device__ __forceinline__ float bf2f(unsigned short h) {
    return __uint_as_float((unsigned int)h << 16);
}

__device__ __forceinline__ void store_bf8(unsigned short* p, const float* v, float s) {
    u32x4 q;
    q[0] = (unsigned int)f2bf(v[0] * s) | ((unsigned int)f2bf(v[1] * s) << 16);
    q[1] = (unsigned int)f2bf(v[2] * s) | ((unsigned int)f2bf(v[3] * s) << 16);
    q[2] = (unsigned int)f2bf(v[4] * s) | ((unsigned int)f2bf(v[5] * s) << 16);
    q[3] = (unsigned int)f2bf(v[6] * s) | ((unsigned int)f2bf(v[7] * s) << 16);
    *(u32x4*)p = q;
}

// ---------------- preprocessing ----------------

// row-wise logsumexp of emit (512 rows x 10000)
__global__ __launch_bounds__(256) void k_row_lse(const float* __restrict__ emit,
                                                 float* __restrict__ lse) {
    __shared__ float sm[256];
    const int n = blockIdx.x, tid = threadIdx.x;
    const float* row = emit + (size_t)n * VOCABN;
    float m = -INFINITY;
    for (int v = tid; v < VOCABN; v += 256) m = fmaxf(m, row[v]);
    sm[tid] = m; __syncthreads();
    for (int s = 128; s > 0; s >>= 1) { if (tid < s) sm[tid] = fmaxf(sm[tid], sm[tid + s]); __syncthreads(); }
    const float M = sm[0]; __syncthreads();
    float su = 0.f;
    for (int v = tid; v < VOCABN; v += 256) su += expf(row[v] - M);
    sm[tid] = su; __syncthreads();
    for (int s = 128; s > 0; s >>= 1) { if (tid < s) sm[tid] += sm[tid + s]; __syncthreads(); }
    if (tid == 0) lse[n] = M + logf(sm[0]);
}

// log_softmax of priors (512)
__global__ __launch_bounds__(512) void k_logpi(const float* __restrict__ priors,
                                               float* __restrict__ logpi) {
    __shared__ float sm[512];
    const int tid = threadIdx.x;
    const float p = priors[tid];
    sm[tid] = p; __syncthreads();
    for (int s = 256; s > 0; s >>= 1) { if (tid < s) sm[tid] = fmaxf(sm[tid], sm[tid + s]); __syncthreads(); }
    const float M = sm[0]; __syncthreads();
    sm[tid] = expf(p - M); __syncthreads();
    for (int s = 256; s > 0; s >>= 1) { if (tid < s) sm[tid] += sm[tid + s]; __syncthreads(); }
    logpi[tid] = p - (M + logf(sm[0]));
}

// column softmax of trans (axis=0) -> bf16 transition-probability matrix P[j,k]
__global__ __launch_bounds__(256) void k_trans(const float* __restrict__ trans,
                                               unsigned short* __restrict__ Pbf) {
    __shared__ float sm[256];
    const int k = blockIdx.x, tid = threadIdx.x;
    const float v0 = trans[(size_t)tid * NSTATES + k];
    const float v1 = trans[(size_t)(tid + 256) * NSTATES + k];
    sm[tid] = fmaxf(v0, v1); __syncthreads();
    for (int s = 128; s > 0; s >>= 1) { if (tid < s) sm[tid] = fmaxf(sm[tid], sm[tid + s]); __syncthreads(); }
    const float M = sm[0]; __syncthreads();
    sm[tid] = expf(v0 - M) + expf(v1 - M); __syncthreads();
    for (int s = 128; s > 0; s >>= 1) { if (tid < s) sm[tid] += sm[tid + s]; __syncthreads(); }
    const float lseC = M + logf(sm[0]);
    Pbf[(size_t)tid * NSTATES + k]         = f2bf(expf(v0 - lseC));
    Pbf[(size_t)(tid + 256) * NSTATES + k] = f2bf(expf(v1 - lseC));
}

// transposed log-emission table: emT[v][j] = bf16(emit[j][v] - lse[j])  (10000 x 512)
// 64x64 tiles through LDS for coalesced read (along v) and write (along j)
__global__ __launch_bounds__(256) void k_emT(const float* __restrict__ emit,
                                             const float* __restrict__ lse,
                                             unsigned short* __restrict__ emT) {
    __shared__ float tile[64][65];
    const int vt = blockIdx.x * 64;
    const int jt = blockIdx.y * 64;
    const int tx = threadIdx.x & 63;
    const int ty = threadIdx.x >> 6;        // 0..3
#pragma unroll
    for (int i = 0; i < 16; ++i) {
        const int j = jt + ty + 4 * i;
        const int v = vt + tx;
        const float val = (v < VOCABN) ? emit[(size_t)j * VOCABN + v] : 0.f;
        tile[ty + 4 * i][tx] = val - lse[j];
    }
    __syncthreads();
#pragma unroll
    for (int i = 0; i < 16; ++i) {
        const int v = vt + ty + 4 * i;
        const int j = jt + tx;
        if (v < VOCABN)
            emT[(size_t)v * NSTATES + j] = f2bf(tile[tx][ty + 4 * i]);
    }
}

// ---------------- main persistent forward kernel ----------------
// grid = 4 blocks (16 batches each), 512 threads = 16 wave32s.
// Wave w computes output-state tiles j = [32w, 32w+32) via WMMA bf16.
template <bool EMT>
__global__ __launch_bounds__(512) void hmm_fwd(
    const int* __restrict__ x, const int* __restrict__ Tlen,
    const float* __restrict__ logpi, const unsigned short* __restrict__ Pbf,
    const float* __restrict__ emit, const float* __restrict__ lse,
    const unsigned short* __restrict__ emT,
    float* __restrict__ out)
{
    __shared__ __align__(16) unsigned short alpha[2][16 * NSTATES]; // [buf][b][k], bf16
    __shared__ float normAcc[2][16];

    const int tid  = threadIdx.x;
    const int wave = tid >> 5;
    const int lane = tid & 31;
    const int half = lane >> 4;
    const int bl   = lane & 15;             // WMMA N index = batch within tile
    const int b    = blockIdx.x * 16 + bl;
    const int j00  = (wave * 2 + 0) * 16;   // A/D tile row bases
    const int j01  = (wave * 2 + 1) * 16;
    const int jr0  = j00 + 8 * half;        // this lane's D rows (M = r + 8*half)
    const int jr1  = j01 + 8 * half;

    if (tid < 32) normAcc[tid >> 4][tid & 15] = 0.f;

    float lse0[8], lse1[8];
#pragma unroll
    for (int r = 0; r < 8; ++r) { lse0[r] = lse[jr0 + r]; lse1[r] = lse[jr1 + r]; }
    const int Tb = Tlen[b];
    __syncthreads();

    // ---- t = 0: alpha0 = exp(em0 + logpi), normalize, logZ = log(norm) ----
    float logZ = 0.f;
    {
        const int tok = x[b * TMAXN + 0];
        float av0[8], av1[8]; float part = 0.f;
#pragma unroll
        for (int r = 0; r < 8; ++r) {
            float e0 = expf(emit[(size_t)(jr0 + r) * VOCABN + tok] - lse0[r] + logpi[jr0 + r]);
            float e1 = expf(emit[(size_t)(jr1 + r) * VOCABN + tok] - lse1[r] + logpi[jr1 + r]);
            av0[r] = e0; av1[r] = e1; part += e0 + e1;
        }
        atomicAdd(&normAcc[0][bl], part);
        __syncthreads();
        const float nrm = normAcc[0][bl];
        logZ = logf(nrm);
        if (wave == 0 && half == 0 && Tb == 1) out[b] = logZ;
        const float inv = 1.f / nrm;
        store_bf8(&alpha[0][bl * NSTATES + jr0], av0, inv);
        store_bf8(&alpha[0][bl * NSTATES + jr1], av1, inv);
        __syncthreads();
    }

    // ---- t = 1 .. TMAX-1 ----
    for (int t = 1; t < TMAXN; ++t) {
        const int s = t & 1;
        const unsigned short* aPrev = &alpha[s ^ 1][0];
        // operand base pointers (16-bit WMMA A/B VGPR layouts per CDNA5 ISA 7.12.2)
        const unsigned short* pB  = aPrev + bl * NSTATES + half * 8;               // B: K = (i/8)*16 + half*8 + i%8
        const unsigned short* pA0 = Pbf + (size_t)(j00 + bl) * NSTATES + half * 8; // A: K = (i/8)*16 + half*8 + i%8
        const unsigned short* pA1 = Pbf + (size_t)(j01 + bl) * NSTATES + half * 8;

        // issue emission loads for this step up front so they overlap the WMMA pipeline
        const int tok = x[b * TMAXN + t];
        float em0[8], em1[8];
        if (EMT) {
            const unsigned short* pe = emT + (size_t)tok * NSTATES;
            const u32x4 q0 = *(const u32x4*)(pe + jr0);   // 8 contiguous bf16
            const u32x4 q1 = *(const u32x4*)(pe + jr1);
#pragma unroll
            for (int r = 0; r < 8; ++r) {
                const unsigned int w0 = q0[r >> 1], w1 = q1[r >> 1];
                em0[r] = bf2f((r & 1) ? (unsigned short)(w0 >> 16) : (unsigned short)(w0 & 0xffffu));
                em1[r] = bf2f((r & 1) ? (unsigned short)(w1 >> 16) : (unsigned short)(w1 & 0xffffu));
            }
        } else {
#pragma unroll
            for (int r = 0; r < 8; ++r) {
                em0[r] = emit[(size_t)(jr0 + r) * VOCABN + tok] - lse0[r];
                em1[r] = emit[(size_t)(jr1 + r) * VOCABN + tok] - lse1[r];
            }
        }

        v8f acc0 = {}; v8f acc1 = {};
#pragma unroll 4
        for (int kc = 0; kc < NSTATES; kc += 32) {
            ABu bb, a0, a1;
            bb.q[0] = *(const u32x4*)(pB + kc);
            bb.q[1] = *(const u32x4*)(pB + kc + 16);
            a0.q[0] = *(const u32x4*)(pA0 + kc);
            a0.q[1] = *(const u32x4*)(pA0 + kc + 16);
            a1.q[0] = *(const u32x4*)(pA1 + kc);
            a1.q[1] = *(const u32x4*)(pA1 + kc + 16);
            acc0 = __builtin_amdgcn_wmma_f32_16x16x32_bf16(false, a0.v, false, bb.v,
                                                           (short)0, acc0, false, false);
            acc1 = __builtin_amdgcn_wmma_f32_16x16x32_bf16(false, a1.v, false, bb.v,
                                                           (short)0, acc1, false, false);
        }

        // emission scale + partial norm
        float av0[8], av1[8]; float part = 0.f;
#pragma unroll
        for (int r = 0; r < 8; ++r) {
            const float v0 = expf(em0[r]) * acc0[r];
            const float v1 = expf(em1[r]) * acc1[r];
            av0[r] = v0; av1[r] = v1; part += v0 + v1;
        }
        atomicAdd(&normAcc[s][bl], part);
        __syncthreads();                       // barrier 1: norm complete

        const float nrm = normAcc[s][bl];
        logZ += logf(nrm);
        if (wave == 0 && half == 0 && t == Tb - 1) out[b] = logZ;
        if (tid < 16) normAcc[s ^ 1][tid] = 0.f;   // recycle slot for step t+1
        const float inv = 1.f / nrm;
        unsigned short* aNext = &alpha[s][0];
        store_bf8(aNext + bl * NSTATES + jr0, av0, inv);
        store_bf8(aNext + bl * NSTATES + jr1, av1, inv);
        __syncthreads();                       // barrier 2: alpha[t] ready
    }
}

extern "C" void kernel_launch(void* const* d_in, const int* in_sizes, int n_in,
                              void* d_out, int out_size, void* d_ws, size_t ws_size,
                              hipStream_t stream) {
    const int*   x      = (const int*)d_in[0];
    const int*   T      = (const int*)d_in[1];
    const float* priors = (const float*)d_in[2];
    const float* trans  = (const float*)d_in[3];
    const float* emit   = (const float*)d_in[4];
    float* out = (float*)d_out;

    float*          lse   = (float*)d_ws;                       // 512 f32
    float*          logpi = (float*)((char*)d_ws + 2048);       // 512 f32
    unsigned short* Pbf   = (unsigned short*)((char*)d_ws + 4096); // 512x512 bf16 = 512KB

    const size_t base_need = 4096 + (size_t)NSTATES * NSTATES * 2;
    const size_t emT_need  = base_need + (size_t)VOCABN * NSTATES * 2; // +10.24MB
    unsigned short* emT = (unsigned short*)((char*)d_ws + base_need);
    const bool useT = (ws_size >= emT_need);

    k_row_lse<<<NSTATES, 256, 0, stream>>>(emit, lse);
    k_logpi  <<<1, NSTATES, 0, stream>>>(priors, logpi);
    k_trans  <<<NSTATES, 256, 0, stream>>>(trans, Pbf);
    if (useT) {
        dim3 g((VOCABN + 63) / 64, NSTATES / 64);
        k_emT<<<g, 256, 0, stream>>>(emit, lse, emT);
        hmm_fwd<true><<<4, 512, 0, stream>>>(x, T, logpi, Pbf, emit, lse, emT, out);
    } else {
        hmm_fwd<false><<<4, 512, 0, stream>>>(x, T, logpi, Pbf, emit, lse, (const unsigned short*)nullptr, out);
    }
}
